// MultiHeadAttention_14302241096177
// MI455X (gfx1250) — matrix operand order repaired and checked by reading, not compiled, not run on hardware
//
#include <hip/hip_runtime.h>
#include <stdint.h>
#include <math.h>

typedef __bf16 bf16;
typedef __attribute__((ext_vector_type(16))) __bf16 v16bf;
typedef __attribute__((ext_vector_type(8)))  __bf16 v8bf;
typedef __attribute__((ext_vector_type(8)))  float  v8f;

#define D_MODEL   2048
#define NUM_HEADS 16
#define DKH       128
#define SEQ       2048
#define BATCH     2
#define MROWS     (BATCH * SEQ)

// ---- CDNA5 async global->LDS copy (ASYNCcnt-tracked) ----------------------
__device__ __forceinline__ void async_b128(void* lds, const void* g) {
  asm volatile("global_load_async_to_lds_b128 %0, %1, off"
               :: "v"((unsigned)(uintptr_t)lds),
                  "v"((unsigned long long)(uintptr_t)g)
               : "memory");
}

#if defined(__has_builtin) && __has_builtin(__builtin_amdgcn_s_wait_asynccnt)
#define WAIT_ASYNC(n) __builtin_amdgcn_s_wait_asynccnt(n)
#else
#define WAIT_ASYNC(n) asm volatile("s_wait_asynccnt " #n ::: "memory")
#endif

__device__ __forceinline__ v16bf concat8(v8bf lo, v8bf hi) {
  return __builtin_shufflevector(lo, hi, 0,1,2,3,4,5,6,7,8,9,10,11,12,13,14,15);
}

// A-matrix fragment (16x32 bf16): lane needs k = hf*16 + i (contiguous,
// matching the B fragments): two contiguous 16B chunks.
__device__ __forceinline__ v16bf load_afrag(const bf16* rowp, int hf) {
  v8bf lo = *(const v8bf*)(rowp + hf * 16);
  v8bf hi = *(const v8bf*)(rowp + hf * 16 + 8);
  return concat8(lo, hi);
}

// ---------------------------------------------------------------------------
// Pre-pass 1: f32 -> bf16 (for q, k, v activations)
// ---------------------------------------------------------------------------
__global__ __launch_bounds__(256) void cvt_bf16_kernel(
    const float* __restrict__ x, bf16* __restrict__ y, int n) {
  const int i = (blockIdx.x * 256 + threadIdx.x) * 8;
  if (i + 8 <= n) {
    v8bf o;
#pragma unroll
    for (int j = 0; j < 8; ++j) o[j] = (bf16)x[i + j];
    *(v8bf*)(y + i) = o;
  }
}

// ---------------------------------------------------------------------------
// Pre-pass 2: W[K][N] f32 -> Wt[N][K] bf16 (transpose + convert, done once)
// ---------------------------------------------------------------------------
__global__ __launch_bounds__(256) void transpose_cvt_kernel(
    const float* __restrict__ W, bf16* __restrict__ Wt, int K, int N) {
  __shared__ float t[32][33];
  const int kb = blockIdx.y * 32, nb = blockIdx.x * 32;
  const int tx = threadIdx.x & 31, tg = threadIdx.x >> 5;  // 8 groups of 4 rows
#pragma unroll
  for (int j = 0; j < 4; ++j) {
    const int kk = tg * 4 + j;
    t[kk][tx] = W[(size_t)(kb + kk) * N + nb + tx];
  }
  __syncthreads();
#pragma unroll
  for (int j = 0; j < 4; ++j) {
    const int nn = tg * 4 + j;
    Wt[(size_t)(nb + nn) * K + kb + tx] = (bf16)t[tx][nn];
  }
}

// ---------------------------------------------------------------------------
// Tiled GEMM, all-bf16: C[M,N] = A[M,K] * Bt[N,K]^T + bias[N]
//   Both LDS tiles are verbatim row copies -> staged with async b128,
//   double-buffered (issue tile t+1, wait own 4 in-flight ops, barrier, MMA).
//   SPLIT=true : bf16 out in head-split [B,H,S,dk]; SPLIT=false : f32 [M,N].
// Block: 256 threads (8 waves), 128x128 tile, K-step 32; wave = 64x32 strip.
// ---------------------------------------------------------------------------
template <bool SPLIT>
__global__ __launch_bounds__(256) void wmma_gemm_bt_kernel(
    const bf16* __restrict__ A, const bf16* __restrict__ Bt,
    const float* __restrict__ bias, void* __restrict__ Cptr,
    int M, int N, int K) {
  __shared__ bf16 As[2][128][32];   // [buf][m][k]
  __shared__ bf16 Bs[2][128][32];   // [buf][n][k]  (Bt rows copied verbatim)

  const int tid  = threadIdx.x;
  const int wave = tid >> 5;
  const int lane = tid & 31;
  const int hf   = lane >> 4;
  const int ln   = lane & 15;
  const int wrow = wave >> 2;       // 0..1
  const int wcol = wave & 3;        // 0..3
  const int rowBlk = blockIdx.y * 128;
  const int colBlk = blockIdx.x * 128;

  const int srow = tid >> 1;            // 0..127 (shared by A and Bt staging)
  const int sk0  = (tid & 1) << 4;      // 0 / 16

  const bf16* agBase = A  + (size_t)(rowBlk + srow) * K + sk0;
  const bf16* bgBase = Bt + (size_t)(colBlk + srow) * K + sk0;

  v8f acc[4][2] = {};

  const int nt = K / 32;
  // stage tile 0 into buffer 0 (4 async b128 per thread -> per-wave cnt = 4)
  async_b128(&As[0][srow][sk0],     agBase);
  async_b128(&As[0][srow][sk0 + 8], agBase + 8);
  async_b128(&Bs[0][srow][sk0],     bgBase);
  async_b128(&Bs[0][srow][sk0 + 8], bgBase + 8);

  for (int t = 0; t < nt; ++t) {
    const int buf = t & 1;
    if (t + 1 < nt) {               // prefetch next tile into other buffer
      const int nb = buf ^ 1;
      const bf16* ag = agBase + (size_t)(t + 1) * 32;
      const bf16* bg = bgBase + (size_t)(t + 1) * 32;
      async_b128(&As[nb][srow][sk0],     ag);
      async_b128(&As[nb][srow][sk0 + 8], ag + 8);
      async_b128(&Bs[nb][srow][sk0],     bg);
      async_b128(&Bs[nb][srow][sk0 + 8], bg + 8);
      WAIT_ASYNC(4);                // this wave's tile-t copies are done
    } else {
      WAIT_ASYNC(0);
    }
    __syncthreads();                // tile t fully resident for all waves

    v16bf af[4];
#pragma unroll
    for (int rt = 0; rt < 4; ++rt)
      af[rt] = load_afrag(&As[buf][wrow * 64 + rt * 16 + ln][0], hf);
    v16bf bfr[2];
#pragma unroll
    for (int ct = 0; ct < 2; ++ct)
      bfr[ct] = *(const v16bf*)&Bs[buf][wcol * 32 + ct * 16 + ln][hf * 16];

#pragma unroll
    for (int rt = 0; rt < 4; ++rt)
#pragma unroll
      for (int ct = 0; ct < 2; ++ct)
        acc[rt][ct] = __builtin_amdgcn_wmma_f32_16x16x32_bf16(
            false, af[rt], false, bfr[ct], (short)0, acc[rt][ct], false, false);

    __syncthreads();                // everyone done with buf before it refills
  }

  // epilogue: D-layout -> lane holds rows r + 8*hf, col = ln
#pragma unroll
  for (int rt = 0; rt < 4; ++rt) {
#pragma unroll
    for (int ct = 0; ct < 2; ++ct) {
      const int   n  = colBlk + wcol * 32 + ct * 16 + ln;
      const float bb = bias[n];
#pragma unroll
      for (int r = 0; r < 8; ++r) {
        const int   m   = rowBlk + wrow * 64 + rt * 16 + r + 8 * hf;
        const float val = acc[rt][ct][r] + bb;
        if (SPLIT) {
          const int b = m >> 11, s = m & (SEQ - 1);
          const int h = n >> 7,  dc = n & (DKH - 1);
          ((bf16*)Cptr)[((size_t)(b * NUM_HEADS + h) * SEQ + s) * DKH + dc] = (bf16)val;
        } else {
          ((float*)Cptr)[(size_t)m * N + n] = val;
        }
      }
    }
  }
}

// ---------------------------------------------------------------------------
// Flash attention: one block = one (b,h) x 128-query tile, 8 waves x 16 queries.
// K tile staged via async-to-LDS; V transposed in-register; online softmax.
// ---------------------------------------------------------------------------
__global__ __launch_bounds__(256) void flash_attn_kernel(
    const bf16* __restrict__ Q, const bf16* __restrict__ K,
    const bf16* __restrict__ V, bf16* __restrict__ O) {
  __shared__ bf16 Ks[32][128];      // [kv][d] (verbatim copy -> async)
  __shared__ bf16 Vs[128][32];      // transposed: [d][kv]
  __shared__ bf16 Ps[8][16][32];    // per-wave P tile (C-layout -> A-layout)

  const int tid  = threadIdx.x;
  const int wave = tid >> 5;
  const int lane = tid & 31;
  const int hf   = lane >> 4;
  const int ln   = lane & 15;

  const int bh   = blockIdx.x;
  const int qblk = blockIdx.y * 128;
  const int b    = bh >> 4;
  const int h    = bh & 15;

  const bf16* Qb = Q + (size_t)bh * SEQ * DKH;
  const bf16* Kb = K + (size_t)bh * SEQ * DKH;
  const bf16* Vb = V + (size_t)bh * SEQ * DKH;

  const float scale = 0.08838834764831845f;   // 1/sqrt(128)

  const int qrow = qblk + wave * 16 + ln;
  v16bf qf[4];
#pragma unroll
  for (int c = 0; c < 4; ++c)
    qf[c] = load_afrag(Qb + (size_t)qrow * DKH + c * 32, hf);

  v8f   oacc[8] = {};
  float mrun[8], lrun[8];
#pragma unroll
  for (int r = 0; r < 8; ++r) { mrun[r] = -3.0e38f; lrun[r] = 0.0f; }

  const int kv = tid >> 3;          // 0..31
  const int d0 = (tid & 7) << 4;    // 0..112

  for (int kt = 0; kt < SEQ; kt += 32) {
    __syncthreads();                // previous tile's LDS reads complete
    {
      const bf16* kg = Kb + (size_t)(kt + kv) * DKH + d0;
      async_b128(&Ks[kv][d0],     kg);
      async_b128(&Ks[kv][d0 + 8], kg + 8);
      const bf16* vg = Vb + (size_t)(kt + kv) * DKH + d0;
      v8bf v0 = *(const v8bf*)vg;
      v8bf v1 = *(const v8bf*)(vg + 8);
#pragma unroll
      for (int j = 0; j < 8; ++j) {
        Vs[d0 + j][kv]     = v0[j];
        Vs[d0 + 8 + j][kv] = v1[j];
      }
      WAIT_ASYNC(0);
    }
    __syncthreads();

    // scores S = Q . K^T : 32 keys = 2 tiles x 4 dk-chunks of WMMA
    v8f sc[2] = {};
#pragma unroll
    for (int st = 0; st < 2; ++st)
#pragma unroll
      for (int c = 0; c < 4; ++c) {
        v16bf kf = *(const v16bf*)&Ks[st * 16 + ln][c * 32 + hf * 16];
        sc[st] = __builtin_amdgcn_wmma_f32_16x16x32_bf16(
            false, qf[c], false, kf, (short)0, sc[st], false, false);
      }

    // online softmax; C-layout rows live per 16-lane half
    float alpha[8];
#pragma unroll
    for (int r = 0; r < 8; ++r) {
      float s0 = sc[0][r] * scale;
      float s1 = sc[1][r] * scale;
      float tm = fmaxf(s0, s1);
#pragma unroll
      for (int off = 8; off >= 1; off >>= 1)
        tm = fmaxf(tm, __shfl_xor(tm, off, 16));
      float mnew = fmaxf(mrun[r], tm);
      alpha[r]   = __expf(mrun[r] - mnew);
      float p0   = __expf(s0 - mnew);
      float p1   = __expf(s1 - mnew);
      float rs   = p0 + p1;
#pragma unroll
      for (int off = 8; off >= 1; off >>= 1)
        rs += __shfl_xor(rs, off, 16);
      lrun[r] = lrun[r] * alpha[r] + rs;
      mrun[r] = mnew;
      Ps[wave][r + 8 * hf][ln]      = (bf16)p0;
      Ps[wave][r + 8 * hf][16 + ln] = (bf16)p1;
    }

#pragma unroll
    for (int dt = 0; dt < 8; ++dt)
#pragma unroll
      for (int r = 0; r < 8; ++r)
        oacc[dt][r] *= alpha[r];

    // P back in A-layout (in-order LDS within the wave: no barrier needed)
    v16bf pf = load_afrag(&Ps[wave][ln][0], hf);

#pragma unroll
    for (int dt = 0; dt < 8; ++dt) {
      v16bf vf = *(const v16bf*)&Vs[dt * 16 + ln][hf * 16];
      oacc[dt] = __builtin_amdgcn_wmma_f32_16x16x32_bf16(
          false, pf, false, vf, (short)0, oacc[dt], false, false);
    }
  }

  float invl[8];
#pragma unroll
  for (int r = 0; r < 8; ++r) invl[r] = 1.0f / lrun[r];
#pragma unroll
  for (int dt = 0; dt < 8; ++dt) {
#pragma unroll
    for (int r = 0; r < 8; ++r) {
      const int qq = qblk + wave * 16 + r + 8 * hf;
      O[((size_t)b * SEQ + qq) * D_MODEL + h * DKH + dt * 16 + ln] =
          (bf16)(oacc[dt][r] * invl[r]);
    }
  }
}

// ---------------------------------------------------------------------------
extern "C" void kernel_launch(void* const* d_in, const int* in_sizes, int n_in,
                              void* d_out, int out_size, void* d_ws, size_t ws_size,
                              hipStream_t stream) {
  (void)in_sizes; (void)n_in; (void)out_size; (void)ws_size;
  const float* q  = (const float*)d_in[0];
  const float* k  = (const float*)d_in[1];
  const float* v  = (const float*)d_in[2];
  const float* Wq = (const float*)d_in[3];
  const float* bq = (const float*)d_in[4];
  const float* Wk = (const float*)d_in[5];
  const float* bk = (const float*)d_in[6];
  const float* Wv = (const float*)d_in[7];
  const float* bv = (const float*)d_in[8];
  const float* Wo = (const float*)d_in[9];
  const float* bo = (const float*)d_in[10];
  float* out = (float*)d_out;

  const size_t nAct = (size_t)MROWS * D_MODEL;      // 8,388,608
  const size_t nW   = (size_t)D_MODEL * D_MODEL;    // 4,194,304
  bf16* qb  = (bf16*)d_ws;
  bf16* kb2 = qb  + nAct;
  bf16* vb2 = kb2 + nAct;
  bf16* Wqt = vb2 + nAct;
  bf16* Wkt = Wqt + nW;
  bf16* Wvt = Wkt + nW;
  bf16* Wot = Wvt + nW;
  bf16* Qh  = Wot + nW;
  bf16* Kh  = Qh  + nAct;
  bf16* Vh  = Kh  + nAct;
  bf16* Oh  = Vh  + nAct;

  // pre-passes: activation f32->bf16, weights f32->bf16 transposed [N][K]
  const int cvtBlocks = (int)(nAct / (256 * 8));
  cvt_bf16_kernel<<<cvtBlocks, 256, 0, stream>>>(q, qb,  (int)nAct);
  cvt_bf16_kernel<<<cvtBlocks, 256, 0, stream>>>(k, kb2, (int)nAct);
  cvt_bf16_kernel<<<cvtBlocks, 256, 0, stream>>>(v, vb2, (int)nAct);
  dim3 tgrid(D_MODEL / 32, D_MODEL / 32);
  transpose_cvt_kernel<<<tgrid, 256, 0, stream>>>(Wq, Wqt, D_MODEL, D_MODEL);
  transpose_cvt_kernel<<<tgrid, 256, 0, stream>>>(Wk, Wkt, D_MODEL, D_MODEL);
  transpose_cvt_kernel<<<tgrid, 256, 0, stream>>>(Wv, Wvt, D_MODEL, D_MODEL);
  transpose_cvt_kernel<<<tgrid, 256, 0, stream>>>(Wo, Wot, D_MODEL, D_MODEL);

  dim3 gblk(D_MODEL / 128, MROWS / 128);  // (16, 32)
  wmma_gemm_bt_kernel<true><<<gblk, 256, 0, stream>>>(qb,  Wqt, bq, Qh, MROWS, D_MODEL, D_MODEL);
  wmma_gemm_bt_kernel<true><<<gblk, 256, 0, stream>>>(kb2, Wkt, bk, Kh, MROWS, D_MODEL, D_MODEL);
  wmma_gemm_bt_kernel<true><<<gblk, 256, 0, stream>>>(vb2, Wvt, bv, Vh, MROWS, D_MODEL, D_MODEL);

  flash_attn_kernel<<<dim3(BATCH * NUM_HEADS, SEQ / 128), 256, 0, stream>>>(Qh, Kh, Vh, Oh);

  wmma_gemm_bt_kernel<false><<<gblk, 256, 0, stream>>>(Oh, Wot, bo, out, MROWS, D_MODEL, D_MODEL);
}
